// Cross_26998164423018
// MI455X (gfx1250) — matrix operand order repaired and checked by reading, not compiled, not run on hardware
//
#include <hip/hip_runtime.h>
#include <hip/hip_bf16.h>

typedef __attribute__((ext_vector_type(16))) _Float16 v16h;
typedef __attribute__((ext_vector_type(8)))  float    v8f;

#define WMMA_F16(a, b, c) \
  __builtin_amdgcn_wmma_f32_16x16x32_f16(false, (a), false, (b), (short)0, (c), false, false)

// ---------------- WMMA fragment loaders (wave32, CDNA5 ISA layouts) ----------------

// A fragment: 16x32 (MxK) f16, source row-major with leading dim `ld` (in halves).
// Lane L (0-15): M=L, K base 0;  Lane L+16: M=L, K base 8.
// VGPR i: K = kb + (i&4 ? 16 : 0) + (i&3)*2  (pair packed lo/hi).
static __device__ __forceinline__ v16h load_a_frag(const _Float16* __restrict__ src,
                                                   int ld, int k0, int lane) {
  const int m  = lane & 15;
  const int kb = k0 + ((lane & 16) ? 8 : 0);
  const _Float16* p = src + (size_t)m * ld;
  v16h a;
#pragma unroll
  for (int i = 0; i < 8; ++i) {
    const int k = kb + ((i & 4) ? 16 : 0) + (i & 3) * 2;
    a[2 * i]     = p[k];
    a[2 * i + 1] = p[k + 1];
  }
  return a;
}

// B fragment: 32x16 (KxN) with B[k][n] = src[n0+n][k0+k]  (src row-major over n; this is
// the torch Linear W[o][c] layout, K^T for Q·K^T, and Vt for P·V). Two 16B runs per lane (K = kb..kb+7, kb+16..kb+23).
static __device__ __forceinline__ v16h load_bT_frag(const _Float16* __restrict__ src,
                                                    int ld, int n0, int k0, int lane) {
  const int n  = n0 + (lane & 15);
  const int kb = k0 + ((lane & 16) ? 8 : 0);
  const _Float16* p = src + (size_t)n * ld + kb;
  v16h b;
#pragma unroll
  for (int i = 0; i < 16; ++i) b[i] = p[(i < 8) ? i : i + 8];
  return b;
}

// ---------------- Kernel 1: LayerNorm (fp32 in -> fp32 copy [optional] + f16 copy) ------

__global__ __launch_bounds__(256) void ln_kernel(const float* __restrict__ in,
                                                 const float* __restrict__ g,
                                                 const float* __restrict__ b,
                                                 float* __restrict__ out32,
                                                 _Float16* __restrict__ out16) {
  __shared__ float red[256];
  const size_t row = blockIdx.x;
  const float* p = in + row * 576;
  const int tid = threadIdx.x;

  float s = 0.f;
  for (int i = tid; i < 576; i += 256) s += p[i];
  red[tid] = s;
  __syncthreads();
  for (int off = 128; off > 0; off >>= 1) {
    if (tid < off) red[tid] += red[tid + off];
    __syncthreads();
  }
  const float mu = red[0] * (1.0f / 576.0f);
  __syncthreads();

  float s2 = 0.f;
  for (int i = tid; i < 576; i += 256) { const float d = p[i] - mu; s2 += d * d; }
  red[tid] = s2;
  __syncthreads();
  for (int off = 128; off > 0; off >>= 1) {
    if (tid < off) red[tid] += red[tid + off];
    __syncthreads();
  }
  const float rs = rsqrtf(red[0] * (1.0f / 576.0f) + 1e-5f);

  for (int i = tid; i < 576; i += 256) {
    const float v = (p[i] - mu) * rs * g[i] + b[i];
    if (out32) out32[row * 576 + i] = v;
    out16[row * 576 + i] = (_Float16)v;
  }
}

// ---------------- Kernel: fp32 -> f16 conversion (weights) ----------------

__global__ void f2h_kernel(const float* __restrict__ in, _Float16* __restrict__ out, int n) {
  const int i = blockIdx.x * blockDim.x + threadIdx.x;
  if (i < n) out[i] = (_Float16)in[i];
}

// ---------------- Kernel 2: projection GEMM  Out[m][o] = sum_c In[m][c]*W[o][c] ----------
// Block: 16 rows of In staged in LDS; 8 waves cover the 36 output n-tiles.
// trV==0: row-major store [m][o].  trV==1: per-batch transposed store Vt[b][o][n]
// (n = m % 2048), so attention's P·V B-fragments become contiguous loads.

__global__ __launch_bounds__(256) void proj_gemm_kernel(const _Float16* __restrict__ In,
                                                        const _Float16* __restrict__ W,
                                                        _Float16* __restrict__ Out,
                                                        int trV) {
  __shared__ _Float16 sIn[16 * 576];
  const int tid = threadIdx.x;
  const size_t row0 = (size_t)blockIdx.x * 16;
  for (int i = tid; i < 16 * 576; i += 256) sIn[i] = In[row0 * 576 + i];
  __syncthreads();

  const int wave = tid >> 5, lane = tid & 31;
  const int colL = lane & 15;
  const int hi   = (lane & 16) ? 8 : 0;

  for (int nt = wave; nt < 36; nt += 8) {
    __builtin_prefetch(W + (size_t)(nt * 16 + colL) * 576, 0, 1);
    v8f acc = {};
    for (int kk = 0; kk < 576; kk += 32) {
      const v16h a = load_a_frag(sIn, 576, kk, lane);
      const v16h b = load_bT_frag(W, 576, nt * 16, kk, lane);
      acc = WMMA_F16(a, b, acc);
    }
    const int col = nt * 16 + colL;
    if (trV == 0) {
#pragma unroll
      for (int r = 0; r < 8; ++r)
        Out[(row0 + r + hi) * 576 + col] = (_Float16)acc[r];
    } else {
      // all 16 rows of this block lie in one batch (2048 % 16 == 0)
      const size_t batch = row0 >> 11;
      const size_t nIn   = (row0 & 2047) + hi;
      _Float16* Ot = Out + batch * (576 * 2048) + (size_t)col * 2048 + nIn;
#pragma unroll
      for (int r = 0; r < 8; ++r) Ot[r] = (_Float16)acc[r];   // 16B contiguous per lane
    }
  }
}

// ---------------- Kernel 3: fused sigmoid attention  O = sigmoid(Q K^T * s) V -----------
// Block: one batch, 32 Q rows (two m-tiles; Q strip staged in LDS, 36 KB).
// Stream K/Vt in 128-column chunks: wave w computes two S tiles for n-tile w (K fragment
// shared across both m-tiles), sigmoid -> P (32x128) in LDS, then all waves accumulate
// O over their d-tiles with each Vt B-fragment feeding two WMMAs. Every fragment load in
// this kernel is two contiguous 16B runs per lane (b128-vectorizable).

__global__ __launch_bounds__(256) void attn_kernel(const _Float16* __restrict__ Q,
                                                   const _Float16* __restrict__ K,
                                                   const _Float16* __restrict__ Vt,
                                                   _Float16* __restrict__ O) {
  __shared__ _Float16 sQ[32 * 576];   // 36 KB
  __shared__ _Float16 sP[32 * 128];   //  8 KB
  const int tid = threadIdx.x;
  const int qt  = blockIdx.x;         // 0..63 (32-row q tile)
  const size_t base = (size_t)blockIdx.y * 2048 * 576;
  const _Float16* Qb  = Q  + base;
  const _Float16* Kb  = K  + base;
  const _Float16* Vtb = Vt + base;    // [576][2048]
  _Float16* Ob = O + base;

  for (int i = tid; i < 32 * 576; i += 256) sQ[i] = Qb[(size_t)qt * 32 * 576 + i];
  __syncthreads();

  const int wave = tid >> 5, lane = tid & 31;
  const int colL = lane & 15;
  const int hi   = (lane & 16) ? 8 : 0;
  // d-tile ownership: wave w owns tiles {w, w+8, w+16, w+24} and (w<4) also w+32 (36 total)
  const int dt0 = wave, dt1 = wave + 8, dt2 = wave + 16, dt3 = wave + 24, dt4 = wave + 32;
  v8f oa0 = {}, oa1 = {}, oa2 = {}, oa3 = {}, oa4 = {};   // m-tile 0 accumulators
  v8f ob0 = {}, ob1 = {}, ob2 = {}, ob3 = {}, ob4 = {};   // m-tile 1 accumulators
  const float scale = 1.0f / 24.0f;   // 576^-0.5

  for (int nc = 0; nc < 2048; nc += 128) {
    // ---- two S tiles (16x16) for key columns [nc + wave*16, +16) ----
    const int n0 = nc + wave * 16;
    __builtin_prefetch(Kb + (size_t)(n0 + colL) * 576, 0, 1);
    v8f s0 = {}, s1 = {};
    for (int kk = 0; kk < 576; kk += 32) {
      const v16h b = load_bT_frag(Kb, 576, n0, kk, lane);          // shared K fragment
      s0 = WMMA_F16(load_a_frag(sQ,            576, kk, lane), b, s0);
      s1 = WMMA_F16(load_a_frag(sQ + 16 * 576, 576, kk, lane), b, s1);
    }
    // ---- sigmoid -> P in LDS (32 x 128) ----
#pragma unroll
    for (int r = 0; r < 8; ++r) {
      const int c = wave * 16 + colL;
      sP[(r + hi) * 128 + c]      = (_Float16)(1.0f / (1.0f + __expf(-s0[r] * scale)));
      sP[(16 + r + hi) * 128 + c] = (_Float16)(1.0f / (1.0f + __expf(-s1[r] * scale)));
    }
    __syncthreads();
    // ---- O += P * V_chunk; each Vt B-fragment feeds both m-tiles ----
#pragma unroll
    for (int ns = 0; ns < 128; ns += 32) {
      const v16h a0 = load_a_frag(sP,            128, ns, lane);
      const v16h a1 = load_a_frag(sP + 16 * 128, 128, ns, lane);
      const v16h b0 = load_bT_frag(Vtb, 2048, dt0 * 16, nc + ns, lane);
      oa0 = WMMA_F16(a0, b0, oa0);  ob0 = WMMA_F16(a1, b0, ob0);
      const v16h b1 = load_bT_frag(Vtb, 2048, dt1 * 16, nc + ns, lane);
      oa1 = WMMA_F16(a0, b1, oa1);  ob1 = WMMA_F16(a1, b1, ob1);
      const v16h b2 = load_bT_frag(Vtb, 2048, dt2 * 16, nc + ns, lane);
      oa2 = WMMA_F16(a0, b2, oa2);  ob2 = WMMA_F16(a1, b2, ob2);
      const v16h b3 = load_bT_frag(Vtb, 2048, dt3 * 16, nc + ns, lane);
      oa3 = WMMA_F16(a0, b3, oa3);  ob3 = WMMA_F16(a1, b3, ob3);
      if (wave < 4) {   // uniform branch: EXEC stays all-ones for WMMA
        const v16h b4 = load_bT_frag(Vtb, 2048, dt4 * 16, nc + ns, lane);
        oa4 = WMMA_F16(a0, b4, oa4);  ob4 = WMMA_F16(a1, b4, ob4);
      }
    }
    __syncthreads();
  }

  const size_t rb = (size_t)qt * 32;
#pragma unroll
  for (int r = 0; r < 8; ++r) {
    const size_t r0 = rb + r + hi;        // m-tile 0 row
    const size_t r1 = rb + 16 + r + hi;   // m-tile 1 row
    Ob[r0 * 576 + dt0 * 16 + colL] = (_Float16)oa0[r];
    Ob[r1 * 576 + dt0 * 16 + colL] = (_Float16)ob0[r];
    Ob[r0 * 576 + dt1 * 16 + colL] = (_Float16)oa1[r];
    Ob[r1 * 576 + dt1 * 16 + colL] = (_Float16)ob1[r];
    Ob[r0 * 576 + dt2 * 16 + colL] = (_Float16)oa2[r];
    Ob[r1 * 576 + dt2 * 16 + colL] = (_Float16)ob2[r];
    Ob[r0 * 576 + dt3 * 16 + colL] = (_Float16)oa3[r];
    Ob[r1 * 576 + dt3 * 16 + colL] = (_Float16)ob3[r];
    if (wave < 4) {
      Ob[r0 * 576 + dt4 * 16 + colL] = (_Float16)oa4[r];
      Ob[r1 * 576 + dt4 * 16 + colL] = (_Float16)ob4[r];
    }
  }
}

// ---------------- Kernel 4: output projection + bias + residual(xn) + LayerNorm ---------

__global__ __launch_bounds__(256) void outproj_ln_kernel(const _Float16* __restrict__ A,
                                                         const _Float16* __restrict__ Wp,
                                                         const float* __restrict__ bp,
                                                         const float* __restrict__ xn,
                                                         const float* __restrict__ gz,
                                                         const float* __restrict__ bz,
                                                         float* __restrict__ out) {
  __shared__ _Float16 sA[16 * 576];    // 18 KB
  __shared__ float    sRow[16 * 576];  // 36 KB (pre-LN rows)
  __shared__ float    sRed[16 * 16];
  __shared__ float    sMu[16], sRs[16];
  const int tid = threadIdx.x;
  const size_t row0 = (size_t)blockIdx.x * 16;
  for (int i = tid; i < 16 * 576; i += 256) sA[i] = A[row0 * 576 + i];
  __syncthreads();

  const int wave = tid >> 5, lane = tid & 31;
  const int colL = lane & 15;
  const int hi   = (lane & 16) ? 8 : 0;

  for (int nt = wave; nt < 36; nt += 8) {
    v8f acc = {};
    for (int kk = 0; kk < 576; kk += 32) {
      const v16h a = load_a_frag(sA, 576, kk, lane);
      const v16h b = load_bT_frag(Wp, 576, nt * 16, kk, lane);
      acc = WMMA_F16(a, b, acc);
    }
    const int col = nt * 16 + colL;
    const float bias = bp[col];
#pragma unroll
    for (int r = 0; r < 8; ++r) {
      const int m = r + hi;
      sRow[m * 576 + col] = acc[r] + bias + xn[(row0 + m) * 576 + col];
    }
  }
  __syncthreads();

  // Per-row LayerNorm: 16 threads per row (256 = 16 rows x 16 slots)
  const int rrow = tid >> 4;
  const int slot = tid & 15;
  float s = 0.f, s2 = 0.f;
  for (int i = slot; i < 576; i += 16) {
    const float v = sRow[rrow * 576 + i];
    s += v; s2 += v * v;
  }
  sRed[rrow * 16 + slot] = s;
  __syncthreads();
  if (slot == 0) {
    float t = 0.f;
    for (int i = 0; i < 16; ++i) t += sRed[rrow * 16 + i];
    sMu[rrow] = t * (1.0f / 576.0f);
  }
  __syncthreads();
  sRed[rrow * 16 + slot] = s2;
  __syncthreads();
  if (slot == 0) {
    float t = 0.f;
    for (int i = 0; i < 16; ++i) t += sRed[rrow * 16 + i];
    const float mu = sMu[rrow];
    sRs[rrow] = rsqrtf(t * (1.0f / 576.0f) - mu * mu + 1e-5f);
  }
  __syncthreads();
  const float mu = sMu[rrow];
  const float rs = sRs[rrow];
  for (int i = slot; i < 576; i += 16)
    out[(row0 + rrow) * 576 + i] = (sRow[rrow * 576 + i] - mu) * rs * gz[i] + bz[i];
}

// ---------------- Host launcher ----------------

extern "C" void kernel_launch(void* const* d_in, const int* in_sizes, int n_in,
                              void* d_out, int out_size, void* d_ws, size_t ws_size,
                              hipStream_t stream) {
  (void)in_sizes; (void)n_in; (void)out_size; (void)ws_size;
  const float* x  = (const float*)d_in[0];
  const float* y  = (const float*)d_in[1];
  const float* Wq = (const float*)d_in[2];
  const float* Wk = (const float*)d_in[3];
  const float* Wv = (const float*)d_in[4];
  const float* Wp = (const float*)d_in[5];
  const float* bp = (const float*)d_in[6];
  const float* gx = (const float*)d_in[7];
  const float* bx = (const float*)d_in[8];
  const float* gy = (const float*)d_in[9];
  const float* by = (const float*)d_in[10];
  const float* gz = (const float*)d_in[11];
  const float* bz = (const float*)d_in[12];
  float* out = (float*)d_out;

  const size_t M = 16384, C = 576, W2 = 576 * 576;
  char* ws = (char*)d_ws;
  size_t off = 0;
  auto alloc = [&](size_t bytes) -> void* {
    void* p = ws + off;
    off = (off + bytes + 255) & ~(size_t)255;
    return p;
  };
  float*    xn32 = (float*)alloc(M * C * 4);
  _Float16* xnh  = (_Float16*)alloc(M * C * 2);
  _Float16* ynh  = (_Float16*)alloc(M * C * 2);
  _Float16* qh   = (_Float16*)alloc(M * C * 2);
  _Float16* kh   = (_Float16*)alloc(M * C * 2);
  _Float16* vth  = (_Float16*)alloc(M * C * 2);   // transposed V: [b][d][n]
  _Float16* aoh  = (_Float16*)alloc(M * C * 2);
  _Float16* wqh  = (_Float16*)alloc(W2 * 2);
  _Float16* wkh  = (_Float16*)alloc(W2 * 2);
  _Float16* wvh  = (_Float16*)alloc(W2 * 2);
  _Float16* wph  = (_Float16*)alloc(W2 * 2);

  // 1) LayerNorms (xn also kept in fp32 for the residual path)
  ln_kernel<<<(unsigned)M, 256, 0, stream>>>(x, gx, bx, xn32, xnh);
  ln_kernel<<<(unsigned)M, 256, 0, stream>>>(y, gy, by, nullptr, ynh);

  // 2) Weight conversion to f16
  const int wn = (int)W2;
  f2h_kernel<<<(wn + 255) / 256, 256, 0, stream>>>(Wq, wqh, wn);
  f2h_kernel<<<(wn + 255) / 256, 256, 0, stream>>>(Wk, wkh, wn);
  f2h_kernel<<<(wn + 255) / 256, 256, 0, stream>>>(Wv, wvh, wn);
  f2h_kernel<<<(wn + 255) / 256, 256, 0, stream>>>(Wp, wph, wn);

  // 3) Q/K/V projections (WMMA); V stored transposed per batch
  proj_gemm_kernel<<<(unsigned)(M / 16), 256, 0, stream>>>(ynh, wqh, qh, 0);
  proj_gemm_kernel<<<(unsigned)(M / 16), 256, 0, stream>>>(xnh, wkh, kh, 0);
  proj_gemm_kernel<<<(unsigned)(M / 16), 256, 0, stream>>>(xnh, wvh, vth, 1);

  // 4) Fused sigmoid attention (WMMA; all fragment loads contiguous; never
  //    materializes the 2048x2048 matrix)
  attn_kernel<<<dim3(64, 8), 256, 0, stream>>>(qh, kh, vth, aoh);

  // 5) Output projection + bias + residual + final LayerNorm
  outproj_ln_kernel<<<(unsigned)(M / 16), 256, 0, stream>>>(aoh, wph, bp, xn32, gz, bz, out);
}
